// HAN_60266981097654
// MI455X (gfx1250) — matrix operand order repaired and checked by reading, not compiled, not run on hardware
//
#include <hip/hip_runtime.h>

// ---------------- problem constants (match reference) ----------------
constexpr int N_NODES = 20000;
constexpr int E_EDGES = 320000;
constexpr int F_IN    = 256;
constexpr int H_HEADS = 8;
constexpr int D_HEAD  = 64;
constexpr int M_PATHS = 2;
constexpr int C_CLS   = 5;
constexpr int HID     = 128;
constexpr int HD      = H_HEADS * D_HEAD;  // 512

typedef __attribute__((ext_vector_type(16))) _Float16 v16h;
typedef __attribute__((ext_vector_type(8)))  float    v8f;

// ---------------- helpers ----------------
__global__ void fill_f32_kernel(float* __restrict__ p, float v, int n) {
    int i = blockIdx.x * blockDim.x + threadIdx.x;
    if (i < n) p[i] = v;
}

// ---------------- operand packing into WMMA fragment order ----------------
// Packed layout: for tile t, lane l (0..31), element i (0..15):
//   out[((t*32 + l)*16) + i]   -> one contiguous 32B block per lane.
// fp32 -> hi/lo f16 split fused in.

// A [Mrows x K] row-major; tile t = mt*(K/32) + kt.
// v16h element i: VGPR i/2, half i%2; K = kt*32 + koff(i/2) + (lane&16 ? 8:0) + i%2
__global__ void pack_a_kernel(const float* __restrict__ A,
                              _Float16* __restrict__ Php, _Float16* __restrict__ Plp,
                              int Mrows, int K) {
    int idx = blockIdx.x * blockDim.x + threadIdx.x;
    if (idx >= Mrows * K) return;
    int i    = idx & 15;
    int lane = (idx >> 4) & 31;
    int tile = idx >> 9;
    int nKt = K >> 5;
    int mt = tile / nKt, kt = tile % nKt;
    int v = i >> 1, pos = i & 1;
    int row = (mt << 4) + (lane & 15);
    int kk  = (kt << 5) + ((v < 4) ? (2 * v) : (16 + 2 * (v - 4))) + ((lane & 16) ? 8 : 0) + pos;
    float x = A[(size_t)row * K + kk];
    _Float16 hi = (_Float16)x;
    Php[idx] = hi;
    Plp[idx] = (_Float16)(x - (float)hi);
}

// B [K x Nc] row-major; tile t = kt*(Nc/16) + nt.
// v16h element i: K = kt*32 + (i<8 ? i : 16+(i-8)) + (lane&16 ? 8:0) ; N = nt*16 + (lane&15)
__global__ void pack_b_kernel(const float* __restrict__ B,
                              _Float16* __restrict__ Php, _Float16* __restrict__ Plp,
                              int K, int Nc) {
    int idx = blockIdx.x * blockDim.x + threadIdx.x;
    if (idx >= K * Nc) return;
    int i    = idx & 15;
    int lane = (idx >> 4) & 31;
    int tile = idx >> 9;
    int ntN = Nc >> 4;
    int kt = tile / ntN, nt = tile % ntN;
    int col = (nt << 4) + (lane & 15);
    int kk  = (kt << 5) + ((i < 8) ? i : (16 + (i - 8))) + ((lane & 16) ? 8 : 0);
    float x = B[(size_t)kk * Nc + col];
    _Float16 hi = (_Float16)x;
    Php[idx] = hi;
    Plp[idx] = (_Float16)(x - (float)hi);
}

// ---------------- WMMA GEMM on packed fragments ----------------
// C = (Ahi+Alo)@(Bhi+Blo) ~= Ahi*Bhi + Ahi*Blo + Alo*Bhi  (f32 accum)
// Each wave computes a 16x64 strip (4 tiles). All 4 B fragment pairs are
// loaded up front, then the 12 WMMAs are issued interleaved across the 4
// independent accumulators to hide WMMA latency / hazard slots.
constexpr int NT = 4;

__global__ void gemm_wmma_packed_kernel(const v16h* __restrict__ Ahp, const v16h* __restrict__ Alp,
                                        const v16h* __restrict__ Bhp, const v16h* __restrict__ Blp,
                                        float* __restrict__ C, int Mrows, int K, int Nc) {
    int lane = threadIdx.x & 31;
    int wave = threadIdx.x >> 5;
    int ntN = Nc >> 4;
    int nStrips = ntN / NT;
    int nKt = K >> 5;
    int strip = blockIdx.x * (blockDim.x >> 5) + wave;
    if (strip >= (Mrows >> 4) * nStrips) return;   // wave-uniform branch
    int mt = strip / nStrips;
    int ns = strip % nStrips;

    const v16h* ap = Ahp + (size_t)mt * nKt * 32 + lane;         // step 32 per kt
    const v16h* alp = Alp + (size_t)mt * nKt * 32 + lane;
    const v16h* bhp = Bhp + ((size_t)ns * NT * 32) + lane;       // step ntN*32 per kt
    const v16h* blp = Blp + ((size_t)ns * NT * 32) + lane;

    v8f c[NT] = {};
    for (int kt = 0; kt < nKt; ++kt) {
        v16h ah = ap[(size_t)kt * 32];
        v16h al = alp[(size_t)kt * 32];
        v16h bh[NT], bl[NT];
#pragma unroll
        for (int t = 0; t < NT; ++t) {
            size_t bidx = (size_t)kt * ntN * 32 + (size_t)t * 32;
            bh[t] = bhp[bidx];
            bl[t] = blp[bidx];
        }
        // interleave independent accumulators between dependent WMMAs
#pragma unroll
        for (int t = 0; t < NT; ++t)
            c[t] = __builtin_amdgcn_wmma_f32_16x16x32_f16(false, ah, false, bh[t], (short)0, c[t], false, false);
#pragma unroll
        for (int t = 0; t < NT; ++t)
            c[t] = __builtin_amdgcn_wmma_f32_16x16x32_f16(false, ah, false, bl[t], (short)0, c[t], false, false);
#pragma unroll
        for (int t = 0; t < NT; ++t)
            c[t] = __builtin_amdgcn_wmma_f32_16x16x32_f16(false, al, false, bh[t], (short)0, c[t], false, false);
    }
    // C/D layout: VGPR r -> row m0+r (lanes 0-15) / m0+8+r (lanes 16-31)
    int row0 = (mt << 4) + ((lane & 16) ? 8 : 0);
    int colBase = (ns * NT) << 4;
#pragma unroll
    for (int t = 0; t < NT; ++t) {
        int col = colBase + (t << 4) + (lane & 15);
#pragma unroll
        for (int r = 0; r < 8; ++r)
            C[(size_t)(row0 + r) * Nc + col] = c[t][r];
    }
}

// ---------------- GAT per-metapath kernels ----------------

// el[n,h] = sum_d feat[n,h,d]*al[h,d]; er likewise. One wave per node.
__global__ void eler_kernel(const float* __restrict__ feat,
                            const float* __restrict__ al,
                            const float* __restrict__ ar,
                            float* __restrict__ el, float* __restrict__ er) {
    int lane = threadIdx.x & 31;
    int n = blockIdx.x * (blockDim.x >> 5) + (threadIdx.x >> 5);
    if (n >= N_NODES) return;
    const float* f = feat + (size_t)n * HD;
#pragma unroll
    for (int h = 0; h < H_HEADS; ++h) {
        int j0 = h * D_HEAD + lane;
        float sl = f[j0] * al[h * D_HEAD + lane] + f[j0 + 32] * al[h * D_HEAD + lane + 32];
        float sr = f[j0] * ar[h * D_HEAD + lane] + f[j0 + 32] * ar[h * D_HEAD + lane + 32];
        for (int o = 16; o > 0; o >>= 1) {
            sl += __shfl_xor(sl, o, 32);
            sr += __shfl_xor(sr, o, 32);
        }
        if (lane == 0) { el[n * H_HEADS + h] = sl; er[n * H_HEADS + h] = sr; }
    }
}

__device__ inline void atomic_max_f32(float* addr, float v) {
    if (v >= 0.f) atomicMax((int*)addr, __float_as_int(v));
    else          atomicMin((unsigned int*)addr, __float_as_uint(v));
}

// pass 1: e = leaky_relu(el[src]+er[dst]); segment max into mx[dst]
__global__ void edge_max_kernel(const int* __restrict__ src, const int* __restrict__ dst,
                                const float* __restrict__ el, const float* __restrict__ er,
                                float* __restrict__ ebuf, float* __restrict__ mx) {
    int t = blockIdx.x * blockDim.x + threadIdx.x;
    if (t >= E_EDGES * H_HEADS) return;
    int e = t >> 3, h = t & 7;
    int s = src[e], d = dst[e];
    float v = el[s * H_HEADS + h] + er[d * H_HEADS + h];
    v = v > 0.f ? v : 0.2f * v;
    ebuf[t] = v;
    atomic_max_f32(mx + d * H_HEADS + h, v);
}

__global__ void fix_mx_kernel(float* __restrict__ mx, int n) {
    int i = blockIdx.x * blockDim.x + threadIdx.x;
    if (i < n && mx[i] == -__builtin_huge_valf()) mx[i] = 0.f;
}

// pass 2: ex = exp(e - mx[dst]); den[dst] += ex; ebuf <- ex
__global__ void edge_exp_kernel(const int* __restrict__ dst,
                                float* __restrict__ ebuf,
                                const float* __restrict__ mx,
                                float* __restrict__ den) {
    int t = blockIdx.x * blockDim.x + threadIdx.x;
    if (t >= E_EDGES * H_HEADS) return;
    int e = t >> 3, h = t & 7;
    int d = dst[e];
    float ex = expf(ebuf[t] - mx[d * H_HEADS + h]);
    ebuf[t] = ex;
    atomicAdd(den + d * H_HEADS + h, ex);
}

// pass 3: out[dst,:] += alpha * feat[src,:]   (one block of 128 per edge, float4)
__global__ void edge_agg_kernel(const int* __restrict__ src, const int* __restrict__ dst,
                                const float* __restrict__ ebuf, const float* __restrict__ den,
                                const float* __restrict__ feat, float* __restrict__ out) {
    int e = blockIdx.x;
    int t = threadIdx.x;       // 0..127
    int j = t * 4;             // 0..508 (h constant within the 4)
    int h = j >> 6;
    int s = src[e], d = dst[e];
    float alpha = ebuf[e * H_HEADS + h] / fmaxf(den[d * H_HEADS + h], 1e-9f);
    const float4 f4 = *(const float4*)(feat + (size_t)s * HD + j);
    float* o = out + (size_t)d * HD + j;
    atomicAdd(o + 0, alpha * f4.x);
    atomicAdd(o + 1, alpha * f4.y);
    atomicAdd(o + 2, alpha * f4.z);
    atomicAdd(o + 3, alpha * f4.w);
}

// z = elu(agg + bias)
__global__ void elu_bias_kernel(float* __restrict__ z, const float* __restrict__ bias) {
    int i = blockIdx.x * blockDim.x + threadIdx.x;
    if (i >= N_NODES * HD) return;
    float x = z[i] + bias[i & (HD - 1)];
    z[i] = x > 0.f ? x : (expf(x) - 1.f);
}

// ---------------- semantic attention ----------------
// wscore[row] = tanh(z[row,:] @ W1 + b1) @ W2 ; one 128-thread block per row
__global__ void sem_kernel(const float* __restrict__ z, const float* __restrict__ W1,
                           const float* __restrict__ b1, const float* __restrict__ W2,
                           float* __restrict__ wscore) {
    __shared__ float zrow[HD];
    __shared__ float red[HID];
    int row = blockIdx.x;
    int t = threadIdx.x;    // 0..127
    const float* zp = z + (size_t)row * HD;
    for (int k = t; k < HD; k += HID) zrow[k] = zp[k];
    __syncthreads();
    float acc = 0.f;
    for (int k = 0; k < HD; ++k) acc += zrow[k] * W1[k * HID + t];  // coalesced W1 column
    red[t] = tanhf(acc + b1[t]) * W2[t];
    __syncthreads();
    for (int s = HID / 2; s > 0; s >>= 1) {
        if (t < s) red[t] += red[t + s];
        __syncthreads();
    }
    if (t == 0) wscore[row] = red[0];
}

__global__ void mean_kernel(const float* __restrict__ wscore, float* __restrict__ acc) {
    int i = blockIdx.x * blockDim.x + threadIdx.x;
    if (i >= M_PATHS * N_NODES) return;
    atomicAdd(acc + i / N_NODES, wscore[i] * (1.0f / N_NODES));
}

__global__ void beta_kernel(const float* __restrict__ acc, float* __restrict__ beta) {
    if (threadIdx.x == 0 && blockIdx.x == 0) {
        float a0 = acc[0], a1 = acc[1];
        float m = fmaxf(a0, a1);
        float e0 = expf(a0 - m), e1 = expf(a1 - m);
        float s = e0 + e1;
        beta[0] = e0 / s;
        beta[1] = e1 / s;
    }
}

// out[n,c] = sum_k (beta0*z0[n,k] + beta1*z1[n,k]) * pW[k,c] + pb[c]; wave per node
__global__ void pred_kernel(const float* __restrict__ z, const float* __restrict__ beta,
                            const float* __restrict__ pW, const float* __restrict__ pb,
                            float* __restrict__ out) {
    int lane = threadIdx.x & 31;
    int n = blockIdx.x * (blockDim.x >> 5) + (threadIdx.x >> 5);
    if (n >= N_NODES) return;
    float b0 = beta[0], b1 = beta[1];
    const float* z0 = z + (size_t)n * HD;
    const float* z1 = z + (size_t)N_NODES * HD + (size_t)n * HD;
    float acc[C_CLS] = {0.f, 0.f, 0.f, 0.f, 0.f};
    for (int k = lane; k < HD; k += 32) {
        float hv = b0 * z0[k] + b1 * z1[k];
#pragma unroll
        for (int c = 0; c < C_CLS; ++c) acc[c] += hv * pW[k * C_CLS + c];
    }
#pragma unroll
    for (int c = 0; c < C_CLS; ++c)
        for (int o = 16; o > 0; o >>= 1) acc[c] += __shfl_xor(acc[c], o, 32);
    if (lane == 0) {
#pragma unroll
        for (int c = 0; c < C_CLS; ++c) out[n * C_CLS + c] = acc[c] + pb[c];
    }
}

// ---------------- launcher ----------------
extern "C" void kernel_launch(void* const* d_in, const int* in_sizes, int n_in,
                              void* d_out, int out_size, void* d_ws, size_t ws_size,
                              hipStream_t stream) {
    const float* h      = (const float*)d_in[0];
    const int*   src    = (const int*)  d_in[1];
    const int*   dst    = (const int*)  d_in[2];
    const float* W      = (const float*)d_in[3];
    const float* attn_l = (const float*)d_in[4];
    const float* attn_r = (const float*)d_in[5];
    const float* bias_g = (const float*)d_in[6];
    const float* sem_W1 = (const float*)d_in[7];
    const float* sem_b1 = (const float*)d_in[8];
    const float* sem_W2 = (const float*)d_in[9];
    const float* pred_W = (const float*)d_in[10];
    const float* pred_b = (const float*)d_in[11];
    float* out = (float*)d_out;

    // workspace arena (all sizes are multiples of 256 bytes)
    char* ws = (char*)d_ws;
    size_t off = 0;
    auto carve = [&](size_t bytes) { char* p = ws + off; off += (bytes + 255) & ~(size_t)255; return p; };
    float*    z       = (float*)   carve((size_t)M_PATHS * N_NODES * HD * 4); // [M,N,HD]
    float*    featbuf = (float*)   carve((size_t)N_NODES * HD * 4);
    _Float16* A_hp    = (_Float16*)carve((size_t)N_NODES * F_IN * 2);         // packed fragments
    _Float16* A_lp    = (_Float16*)carve((size_t)N_NODES * F_IN * 2);
    _Float16* B_hp    = (_Float16*)carve((size_t)M_PATHS * F_IN * HD * 2);
    _Float16* B_lp    = (_Float16*)carve((size_t)M_PATHS * F_IN * HD * 2);
    float*    el      = (float*)   carve((size_t)N_NODES * H_HEADS * 4);
    float*    er      = (float*)   carve((size_t)N_NODES * H_HEADS * 4);
    float*    mx      = (float*)   carve((size_t)N_NODES * H_HEADS * 4);
    float*    den     = (float*)   carve((size_t)N_NODES * H_HEADS * 4);
    float*    ebuf    = (float*)   carve((size_t)E_EDGES * H_HEADS * 4);
    float*    wscore  = (float*)   carve((size_t)M_PATHS * N_NODES * 4);
    float*    acc     = (float*)   carve(256);
    float*    beta    = (float*)   carve(256);
    (void)ws_size; (void)n_in; (void)in_sizes; (void)out_size;

    const int TB = 256;
    auto cdiv = [](int a, int b) { return (a + b - 1) / b; };

    // pack A (h) and B (both W matrices) into fragment order, fused hi/lo split
    pack_a_kernel<<<cdiv(N_NODES * F_IN, TB), TB, 0, stream>>>(h, A_hp, A_lp, N_NODES, F_IN);
    for (int m = 0; m < M_PATHS; ++m)
        pack_b_kernel<<<cdiv(F_IN * HD, TB), TB, 0, stream>>>(
            W + (size_t)m * F_IN * HD, B_hp + (size_t)m * F_IN * HD, B_lp + (size_t)m * F_IN * HD,
            F_IN, HD);

    // zero aggregation output z
    fill_f32_kernel<<<cdiv(M_PATHS * N_NODES * HD, TB), TB, 0, stream>>>(z, 0.f, M_PATHS * N_NODES * HD);

    for (int m = 0; m < M_PATHS; ++m) {
        float* zm = z + (size_t)m * N_NODES * HD;

        // feat = h @ W[m]   (WMMA, 16x64 strip per wave, 8 waves/block)
        int nStrips = (N_NODES / 16) * ((HD / 16) / NT);      // 1250 * 8 = 10000
        gemm_wmma_packed_kernel<<<cdiv(nStrips, 8), 256, 0, stream>>>(
            (const v16h*)A_hp, (const v16h*)A_lp,
            (const v16h*)(B_hp + (size_t)m * F_IN * HD), (const v16h*)(B_lp + (size_t)m * F_IN * HD),
            featbuf, N_NODES, F_IN, HD);

        // el/er
        eler_kernel<<<cdiv(N_NODES, 8), 256, 0, stream>>>(
            featbuf, attn_l + m * H_HEADS * D_HEAD, attn_r + m * H_HEADS * D_HEAD, el, er);

        // edge softmax
        fill_f32_kernel<<<cdiv(N_NODES * H_HEADS, TB), TB, 0, stream>>>(mx, -__builtin_huge_valf(), N_NODES * H_HEADS);
        fill_f32_kernel<<<cdiv(N_NODES * H_HEADS, TB), TB, 0, stream>>>(den, 0.f, N_NODES * H_HEADS);
        edge_max_kernel<<<cdiv(E_EDGES * H_HEADS, TB), TB, 0, stream>>>(
            src + (size_t)m * E_EDGES, dst + (size_t)m * E_EDGES, el, er, ebuf, mx);
        fix_mx_kernel<<<cdiv(N_NODES * H_HEADS, TB), TB, 0, stream>>>(mx, N_NODES * H_HEADS);
        edge_exp_kernel<<<cdiv(E_EDGES * H_HEADS, TB), TB, 0, stream>>>(
            dst + (size_t)m * E_EDGES, ebuf, mx, den);

        // scatter-aggregate into z[m]
        edge_agg_kernel<<<E_EDGES, 128, 0, stream>>>(
            src + (size_t)m * E_EDGES, dst + (size_t)m * E_EDGES, ebuf, den, featbuf, zm);

        // bias + elu
        elu_bias_kernel<<<cdiv(N_NODES * HD, TB), TB, 0, stream>>>(zm, bias_g + m * HD);
    }

    // semantic attention
    sem_kernel<<<M_PATHS * N_NODES, HID, 0, stream>>>(z, sem_W1, sem_b1, sem_W2, wscore);
    fill_f32_kernel<<<1, 32, 0, stream>>>(acc, 0.f, 2);
    mean_kernel<<<cdiv(M_PATHS * N_NODES, TB), TB, 0, stream>>>(wscore, acc);
    beta_kernel<<<1, 32, 0, stream>>>(acc, beta);

    // prediction head
    pred_kernel<<<cdiv(N_NODES, 8), 256, 0, stream>>>(z, beta, pred_W, pred_b, out);
}
